// AttentionLayer_42803644072573
// MI455X (gfx1250) — matrix operand order repaired and checked by reading, not compiled, not run on hardware
//
#include <hip/hip_runtime.h>

#define NNODES 1024
#define NE     32768

typedef __attribute__((ext_vector_type(16))) __bf16 v16bf;
typedef __attribute__((ext_vector_type(8)))  __bf16 v8bf;
typedef __attribute__((ext_vector_type(8)))  float  v8f;

static __device__ __forceinline__ float rcp_f(float x){ return __builtin_amdgcn_rcpf(x); }
static __device__ __forceinline__ float silu_f(float x){ return x * rcp_f(1.f + __expf(-x)); }
static __device__ __forceinline__ float sigm_f(float x){ return rcp_f(1.f + __expf(-x)); }

static __device__ __forceinline__ unsigned pk2bf(float a, float b){
  unsigned short ua = __builtin_bit_cast(unsigned short, (__bf16)a);
  unsigned short ub = __builtin_bit_cast(unsigned short, (__bf16)b);
  return (unsigned)ua | ((unsigned)ub << 16);
}
// order-preserving float->uint key for atomic max (init 0 < key(-inf))
static __device__ __forceinline__ unsigned fkey(float f){
  unsigned b = __float_as_uint(f);
  return (b & 0x80000000u) ? ~b : (b | 0x80000000u);
}
static __device__ __forceinline__ float funkey(unsigned e){
  unsigned b = (e & 0x80000000u) ? (e ^ 0x80000000u) : ~e;
  return __uint_as_float(b);
}

// A-tile (16x32 bf16) per ISA 7.12.2: lane<16 row=lane K{kb..kb+7, kb+16..kb+23};
// lane>=16 same row, K{kb+8..15, kb+24..31}. Two ds_load_b128 per lane.
static __device__ __forceinline__ v16bf ldA(const __bf16* base, int stride, int kbase, int lane){
  int r = lane & 15;
  int koff = kbase + ((lane >> 4) << 3);
  const __bf16* p = base + r * stride + koff;
  v8bf x0 = *(const v8bf*)p;
  v8bf x1 = *(const v8bf*)(p + 16);
  v16bf o;
#pragma unroll
  for (int i = 0; i < 8; ++i){ o[i] = x0[i]; o[i + 8] = x1[i]; }
  return o;
}
// B-tile (32x16 bf16): lane = column n; lanes0-15 K{kb..kb+7, kb+16..kb+23}, lanes16-31 K{kb+8..15, kb+24..31}.
// Weights pre-stored transposed [N][Kpad] so this is two 16B loads.
static __device__ __forceinline__ v16bf ldB(const __bf16* Wt, int kstride, int ncol, int kbase, int lane){
  int n = ncol + (lane & 15);
  int k = kbase + ((lane >> 4) << 3);
  const __bf16* p = Wt + n * kstride + k;
  v8bf x0 = *(const v8bf*)p;
  v8bf x1 = *(const v8bf*)(p + 16);
  v16bf o;
#pragma unroll
  for (int i = 0; i < 8; ++i){ o[i] = x0[i]; o[i + 8] = x1[i]; }
  return o;
}
static __device__ __forceinline__ v8f wmma_bf(v16bf a, v16bf b, v8f c){
  return __builtin_amdgcn_wmma_f32_16x16x32_bf16(false, a, false, b, (short)0, c, false, false);
}
// C layout: row = r + 8*(lane>>4), col = lane&15

// ---------------------------------------------------------------- utility
__global__ void k_zero(float* p, int n){
  int t = blockIdx.x * 256 + threadIdx.x;
  if (t < n) p[t] = 0.f;
}

// W [K][N] f32 row-major -> Wt [Npad][Kpad] bf16 (zero padded)
__global__ void k_transpose_bf16(const float* __restrict__ src, __bf16* __restrict__ dst,
                                 int K, int N, int Kpad, int Npad){
  int t = blockIdx.x * 256 + threadIdx.x;
  if (t >= Kpad * Npad) return;
  int n = t / Kpad, k = t - n * Kpad;
  float v = (k < K && n < N) ? src[k * N + n] : 0.f;
  dst[t] = (__bf16)v;
}

// ---------------------------------------------------------------- L/R = left_z/right_z MLP(H)
__global__ __launch_bounds__(64)
void k_leftright(const float* __restrict__ H,
                 const __bf16* LW1t, const __bf16* LW2t, const float* lb1, const float* lb2,
                 const __bf16* RW1t, const __bf16* RW2t, const float* rb1, const float* rb2,
                 float* __restrict__ Lz, float* __restrict__ Rz){
  __shared__ __bf16 sh [2][16][64];
  __shared__ __bf16 sh1[2][16][64];
  int w = threadIdx.x >> 5, lane = threadIdx.x & 31;
  int n0 = blockIdx.x * 32 + w * 16;
  int crow = (lane >> 4) << 3, ccol = lane & 15;
  { // stage H tile -> bf16 LDS
    int r = lane >> 1, half = (lane & 1) * 32;
    const float* ps = H + (n0 + r) * 64 + half;
#pragma unroll
    for (int i = 0; i < 8; ++i){
      float4 v = *(const float4*)(ps + i * 4);
      *(uint2*)&sh[w][r][half + i * 4] = make_uint2(pk2bf(v.x, v.y), pk2bf(v.z, v.w));
    }
  }
  __syncthreads();
  for (int pass = 0; pass < 2; ++pass){
    const __bf16* W1t = pass ? RW1t : LW1t;
    const __bf16* W2t = pass ? RW2t : LW2t;
    const float*  b1  = pass ? rb1 : lb1;
    const float*  b2  = pass ? rb2 : lb2;
    float* dst        = pass ? Rz : Lz;
    v16bf a0 = ldA(&sh[w][0][0], 64, 0, lane);
    v16bf a1 = ldA(&sh[w][0][0], 64, 32, lane);
    for (int nt = 0; nt < 4; ++nt){
      v8f c = {};
      c = wmma_bf(a0, ldB(W1t, 64, nt * 16, 0,  lane), c);
      c = wmma_bf(a1, ldB(W1t, 64, nt * 16, 32, lane), c);
#pragma unroll
      for (int r = 0; r < 8; ++r){
        int row = crow + r, col = nt * 16 + ccol;
        sh1[w][row][col] = (__bf16)silu_f(c[r] + b1[col]);
      }
    }
    __syncthreads();
    a0 = ldA(&sh1[w][0][0], 64, 0, lane);
    a1 = ldA(&sh1[w][0][0], 64, 32, lane);
    for (int nt = 0; nt < 4; ++nt){
      v8f c = {};
      c = wmma_bf(a0, ldB(W2t, 64, nt * 16, 0,  lane), c);
      c = wmma_bf(a1, ldB(W2t, 64, nt * 16, 32, lane), c);
#pragma unroll
      for (int r = 0; r < 8; ++r){
        int row = crow + r, col = nt * 16 + ccol;
        dst[(n0 + row) * 64 + col] = c[r] + b2[col];
      }
    }
    __syncthreads();
  }
}

// ---------------------------------------------------------------- main edge kernel:
// Q MLP, KV MLP (K,V), gathered pair_bias(Z[src,dst]), gate, scores, atomic max.
__global__ __launch_bounds__(64)
void k_edge_main(const float* __restrict__ X, const float* __restrict__ H,
                 const float* __restrict__ E, const int* __restrict__ eidx,
                 const float* __restrict__ Z,
                 const float* kvW1, const float* kvb1, const float* kvb2,
                 const float* qb1, const float* qb2,
                 const float* pbW1, const float* pbb1, const float* pbW2, const float* pbb2,
                 const float* gW1, const float* gb1, const float* gW2, const float* gb2,
                 const __bf16* QW1t, const __bf16* QW2t,
                 const __bf16* KVW1t, const __bf16* KVW2t,
                 float* __restrict__ Vws, float* __restrict__ scores,
                 float* __restrict__ gateb, float* __restrict__ Xrel,
                 unsigned* __restrict__ smaxU){
  __shared__ __bf16 s_ins [2][16][64];
  __shared__ __bf16 s_ind [2][16][64];
  __shared__ __bf16 s_h1kv[2][16][128];
  __shared__ __bf16 s_h1q [2][16][64];
  __shared__ float  s_q[2][16][64];
  __shared__ float  s_k[2][16][64];
  __shared__ int    s_src[2][16], s_dst[2][16];
  __shared__ float  s_rd[2][16];
  int w = threadIdx.x >> 5, lane = threadIdx.x & 31;
  int e0 = blockIdx.x * 32 + w * 16;
  int crow = (lane >> 4) << 3, ccol = lane & 15;

  if (lane < 16){
    int e = e0 + lane;
    int s = eidx[e], d = eidx[NE + e];
    s_src[w][lane] = s; s_dst[w][lane] = d;
    float rx = X[s*3+0]-X[d*3+0], ry = X[s*3+1]-X[d*3+1], rz = X[s*3+2]-X[d*3+2];
    float rd = rx*rx + ry*ry + rz*rz;
    s_rd[w][lane] = rd;
    float inv = rcp_f(1.f + sqrtf(rd + 1e-8f));
    Xrel[e*3+0] = rx*inv; Xrel[e*3+1] = ry*inv; Xrel[e*3+2] = rz*inv;
  }
  __syncthreads();
  { // stage H[src], H[dst] -> bf16
    int r = lane >> 1, half = (lane & 1) * 32;
    const float* ps = H + s_src[w][r] * 64 + half;
    const float* pd = H + s_dst[w][r] * 64 + half;
#pragma unroll
    for (int i = 0; i < 8; ++i){
      float4 a = *(const float4*)(ps + i*4);
      float4 b = *(const float4*)(pd + i*4);
      *(uint2*)&s_ins[w][r][half + i*4] = make_uint2(pk2bf(a.x,a.y), pk2bf(a.z,a.w));
      *(uint2*)&s_ind[w][r][half + i*4] = make_uint2(pk2bf(b.x,b.y), pk2bf(b.z,b.w));
    }
  }
  __syncthreads();
  // KV layer1: [16,64]@[64,128]; K=65 handled as K=64 GEMM + rank-1 (rel_dist * W1row0)
  {
    v16bf a0 = ldA(&s_ind[w][0][0], 64, 0, lane);
    v16bf a1 = ldA(&s_ind[w][0][0], 64, 32, lane);
    for (int nt = 0; nt < 8; ++nt){
      v8f c = {};
      c = wmma_bf(a0, ldB(KVW1t, 64, nt*16, 0,  lane), c);
      c = wmma_bf(a1, ldB(KVW1t, 64, nt*16, 32, lane), c);
#pragma unroll
      for (int r = 0; r < 8; ++r){
        int row = crow + r, col = nt*16 + ccol;
        float pre = c[r] + kvb1[col] + s_rd[w][row] * kvW1[col];
        s_h1kv[w][row][col] = (__bf16)silu_f(pre);
      }
    }
  }
  // Q layer1
  {
    v16bf a0 = ldA(&s_ins[w][0][0], 64, 0, lane);
    v16bf a1 = ldA(&s_ins[w][0][0], 64, 32, lane);
    for (int nt = 0; nt < 4; ++nt){
      v8f c = {};
      c = wmma_bf(a0, ldB(QW1t, 64, nt*16, 0,  lane), c);
      c = wmma_bf(a1, ldB(QW1t, 64, nt*16, 32, lane), c);
#pragma unroll
      for (int r = 0; r < 8; ++r){
        int row = crow + r, col = nt*16 + ccol;
        s_h1q[w][row][col] = (__bf16)silu_f(c[r] + qb1[col]);
      }
    }
  }
  __syncthreads();
  // hoisted A tiles of both hidden layers (reused across all column tiles / heads)
  v16bf akv[4];
#pragma unroll
  for (int i = 0; i < 4; ++i) akv[i] = ldA(&s_h1kv[w][0][0], 128, i*32, lane);
  v16bf aq0 = ldA(&s_h1q[w][0][0], 64, 0, lane);
  v16bf aq1 = ldA(&s_h1q[w][0][0], 64, 32, lane);
  // V = KV layer2 cols 256..511 -> ws
  for (int nt = 0; nt < 16; ++nt){
    v8f c = {};
#pragma unroll
    for (int ks = 0; ks < 4; ++ks)
      c = wmma_bf(akv[ks], ldB(KVW2t, 128, 256 + nt*16, ks*32, lane), c);
#pragma unroll
    for (int r = 0; r < 8; ++r){
      int row = crow + r, col = nt*16 + ccol;
      Vws[(size_t)(e0 + row) * 256 + col] = c[r] + kvb2[256 + col];
    }
  }
  // per-head K,Q tiles -> LDS, dot for scores
  float sc[4];
  for (int h = 0; h < 4; ++h){
    for (int nt = 0; nt < 4; ++nt){
      v8f c = {};
#pragma unroll
      for (int ks = 0; ks < 4; ++ks)
        c = wmma_bf(akv[ks], ldB(KVW2t, 128, h*64 + nt*16, ks*32, lane), c);
      v8f cq = {};
      cq = wmma_bf(aq0, ldB(QW2t, 64, h*64 + nt*16, 0,  lane), cq);
      cq = wmma_bf(aq1, ldB(QW2t, 64, h*64 + nt*16, 32, lane), cq);
#pragma unroll
      for (int r = 0; r < 8; ++r){
        int row = crow + r, col = nt*16 + ccol;
        s_k[w][row][col] = c[r]  + kvb2[h*64 + col];
        s_q[w][row][col] = cq[r] + qb2 [h*64 + col];
      }
    }
    __syncthreads();
    if (lane < 16){
      float acc = 0.f;
      for (int k = 0; k < 64; ++k) acc += s_q[w][lane][k] * s_k[w][lane][k];
      sc[h] = acc * 0.125f; // 1/sqrt(64)
    }
    __syncthreads();
  }
  // pair_bias on gathered Z[src,dst]: lane pair (r=lane>>1, p=lane&1) splits hidden
  float bia[4] = {0.f,0.f,0.f,0.f};
  {
    int r = lane >> 1, p = lane & 1;
    int s = s_src[w][r], d = s_dst[w][r];
    const float* zr = Z + ((size_t)s * 1024 + d) * 64;
    float acc[32];
#pragma unroll
    for (int j = 0; j < 32; ++j) acc[j] = pbb1[p*32 + j];
    for (int k = 0; k < 64; ++k){
      float zv = zr[k];
      const float* wr = pbW1 + k*64 + p*32;
#pragma unroll
      for (int j = 0; j < 32; ++j) acc[j] += zv * wr[j];
    }
#pragma unroll
    for (int j = 0; j < 32; ++j){
      float hj = silu_f(acc[j]);
      const float* w2 = pbW2 + (p*32 + j) * 4;
      bia[0] += hj*w2[0]; bia[1] += hj*w2[1]; bia[2] += hj*w2[2]; bia[3] += hj*w2[3];
    }
  }
#pragma unroll
  for (int o = 0; o < 4; ++o) bia[o] += __shfl_xor(bia[o], 1);
  // gate MLP on E
  float gt[4] = {0.f,0.f,0.f,0.f};
  {
    int r = lane >> 1, p = lane & 1;
    int e = e0 + r;
    float acc[16];
#pragma unroll
    for (int j = 0; j < 16; ++j) acc[j] = gb1[p*16 + j];
    for (int k = 0; k < 32; ++k){
      float ev = E[e*32 + k];
      const float* wr = gW1 + k*32 + p*16;
#pragma unroll
      for (int j = 0; j < 16; ++j) acc[j] += ev * wr[j];
    }
#pragma unroll
    for (int j = 0; j < 16; ++j){
      float hj = silu_f(acc[j]);
      const float* w2 = gW2 + (p*16 + j) * 4;
      gt[0] += hj*w2[0]; gt[1] += hj*w2[1]; gt[2] += hj*w2[2]; gt[3] += hj*w2[3];
    }
  }
#pragma unroll
  for (int o = 0; o < 4; ++o) gt[o] += __shfl_xor(gt[o], 1);
  if ((lane & 1) == 0){
    int e = e0 + (lane >> 1);
#pragma unroll
    for (int o = 0; o < 4; ++o) gateb[e*4 + o] = sigm_f(gt[o] + gb2[o]);
  }
  // scores + atomic segment-max over src
#pragma unroll
  for (int o = 0; o < 4; ++o){
    float bo = __shfl(bia[o], (lane & 15) * 2);
    if (lane < 16){
      int e = e0 + lane;
      float sco = sc[o] + bo + pbb2[o];
      scores[e*4 + o] = sco;
      atomicMax(&smaxU[s_src[w][lane]*4 + o], fkey(sco));
    }
  }
}

// ---------------------------------------------------------------- exp + denom
__global__ void k_exp(const float* __restrict__ scores, const unsigned* __restrict__ smaxU,
                      const int* __restrict__ eidx, float* __restrict__ aij,
                      float* __restrict__ denom){
  int t = blockIdx.x * 256 + threadIdx.x;
  if (t >= NE * 4) return;
  int e = t >> 2, h = t & 3;
  int s = eidx[e];
  float ex = __expf(scores[t] - funkey(smaxU[s*4 + h]));
  aij[t] = ex;
  atomicAdd(&denom[s*4 + h], ex);
}

// ---------------------------------------------------------------- phi_x(V) -> coor_wij
__global__ __launch_bounds__(64)
void k_phix(const float* __restrict__ Vws, const __bf16* W1t, const float* b1,
            const float* W2, const float* b2, float* __restrict__ coorw){
  __shared__ __bf16 sv [2][16][256];
  __shared__ float  sh1[2][16][64];
  int w = threadIdx.x >> 5, lane = threadIdx.x & 31;
  int e0 = blockIdx.x * 32 + w * 16;
  int crow = (lane >> 4) << 3, ccol = lane & 15;
  {
    int r = lane >> 1, half = (lane & 1) * 128;
    const float* ps = Vws + (size_t)(e0 + r) * 256 + half;
#pragma unroll
    for (int i = 0; i < 32; ++i){
      float4 v = *(const float4*)(ps + i*4);
      *(uint2*)&sv[w][r][half + i*4] = make_uint2(pk2bf(v.x,v.y), pk2bf(v.z,v.w));
    }
  }
  __syncthreads();
  v16bf a[8];
#pragma unroll
  for (int i = 0; i < 8; ++i) a[i] = ldA(&sv[w][0][0], 256, i*32, lane);
  for (int nt = 0; nt < 4; ++nt){
    v8f c = {};
#pragma unroll
    for (int ks = 0; ks < 8; ++ks)
      c = wmma_bf(a[ks], ldB(W1t, 256, nt*16, ks*32, lane), c);
#pragma unroll
    for (int r = 0; r < 8; ++r){
      int row = crow + r, col = nt*16 + ccol;
      sh1[w][row][col] = silu_f(c[r] + b1[col]);
    }
  }
  __syncthreads();
  if (lane < 16){
    float acc = b2[0];
    for (int k = 0; k < 64; ++k) acc += sh1[w][lane][k] * W2[k];
    coorw[e0 + lane] = fminf(10.f, fmaxf(-10.f, acc));
  }
}

// ---------------------------------------------------------------- per-edge finalize:
// alpha_ij, alpha(node), x_update scatter, phi_e -> E_out
__global__ void k_edge_final(const int* __restrict__ eidx, const float* __restrict__ denom,
                             const float* __restrict__ gateb, const float* __restrict__ coorw,
                             const float* __restrict__ Xrel, const float* __restrict__ E,
                             const float* __restrict__ H,
                             const float* peW1, const float* peb1,
                             const float* peW2, const float* peb2,
                             float* __restrict__ aij, float* __restrict__ alphaN,
                             float* __restrict__ xupd, float* __restrict__ Eout){
  int e = blockIdx.x * 256 + threadIdx.x;
  if (e >= NE) return;
  int s = eidx[e], d = eidx[NE + e];
  float a4[4], asum = 0.f;
#pragma unroll
  for (int h = 0; h < 4; ++h){
    float al = gateb[e*4+h] * aij[e*4+h] * rcp_f(denom[s*4+h]);
    a4[h] = al; aij[e*4+h] = al; asum += al;
  }
  float am = asum * 0.25f;
  atomicAdd(&alphaN[d], am);
  float cw = coorw[e];
  float xr[3];
#pragma unroll
  for (int c = 0; c < 3; ++c){ xr[c] = Xrel[e*3+c]; atomicAdd(&xupd[d*4+c], am * xr[c] * cw); }
  // phi_e: in = [alpha_ij(4), X_rel_norm(3), H[src](64), H[dst](64)]
  float h1[32];
#pragma unroll
  for (int j = 0; j < 32; ++j) h1[j] = peb1[j];
#pragma unroll
  for (int k = 0; k < 4; ++k){
    const float* wr = peW1 + k*32; float v = a4[k];
#pragma unroll
    for (int j = 0; j < 32; ++j) h1[j] += v * wr[j];
  }
#pragma unroll
  for (int k = 0; k < 3; ++k){
    const float* wr = peW1 + (4+k)*32; float v = xr[k];
#pragma unroll
    for (int j = 0; j < 32; ++j) h1[j] += v * wr[j];
  }
  const float* Hs = H + (size_t)s * 64;
  for (int k = 0; k < 64; ++k){
    const float* wr = peW1 + (7+k)*32; float v = Hs[k];
#pragma unroll
    for (int j = 0; j < 32; ++j) h1[j] += v * wr[j];
  }
  const float* Hd = H + (size_t)d * 64;
  for (int k = 0; k < 64; ++k){
    const float* wr = peW1 + (71+k)*32; float v = Hd[k];
#pragma unroll
    for (int j = 0; j < 32; ++j) h1[j] += v * wr[j];
  }
  float out[32];
#pragma unroll
  for (int o = 0; o < 32; ++o) out[o] = peb2[o];
#pragma unroll
  for (int j = 0; j < 32; ++j){
    float hj = silu_f(h1[j]);
    const float* w2 = peW2 + j*32;
#pragma unroll
    for (int o = 0; o < 32; ++o) out[o] += hj * w2[o];
  }
  for (int o = 0; o < 32; ++o) Eout[e*32 + o] = E[e*32 + o] + out[o];
}

// ---------------------------------------------------------------- attn_out scatter (float4 per thread)
__global__ void k_scatter_attn(const float* __restrict__ aij, const float* __restrict__ Vws,
                               const int* __restrict__ eidx, float* __restrict__ attnO){
  int t = blockIdx.x * 256 + threadIdx.x; // NE*64 total
  int e = t >> 6, k4 = (t & 63) * 4;
  int d = eidx[NE + e];
  float a = aij[e*4 + (k4 >> 6)];
  float4 v = *(const float4*)(Vws + (size_t)e*256 + k4);
  float* p = attnO + d*256 + k4;
  atomicAdd(p + 0, a * v.x);
  atomicAdd(p + 1, a * v.y);
  atomicAdd(p + 2, a * v.z);
  atomicAdd(p + 3, a * v.w);
}

// ---------------------------------------------------------------- H_out = H + phi_h(attn_out)
__global__ __launch_bounds__(64)
void k_phih(const float* __restrict__ attnO, const float* __restrict__ H,
            const __bf16* W1t, const __bf16* W2t, const float* b1, const float* b2,
            float* __restrict__ Hout){
  __shared__ __bf16 sa [2][16][256];
  __shared__ __bf16 sh1[2][16][64];
  int w = threadIdx.x >> 5, lane = threadIdx.x & 31;
  int n0 = blockIdx.x * 32 + w * 16;
  int crow = (lane >> 4) << 3, ccol = lane & 15;
  {
    int r = lane >> 1, half = (lane & 1) * 128;
    const float* ps = attnO + (n0 + r) * 256 + half;
#pragma unroll
    for (int i = 0; i < 32; ++i){
      float4 v = *(const float4*)(ps + i*4);
      *(uint2*)&sa[w][r][half + i*4] = make_uint2(pk2bf(v.x,v.y), pk2bf(v.z,v.w));
    }
  }
  __syncthreads();
  v16bf a[8];
#pragma unroll
  for (int i = 0; i < 8; ++i) a[i] = ldA(&sa[w][0][0], 256, i*32, lane);
  for (int nt = 0; nt < 4; ++nt){
    v8f c = {};
#pragma unroll
    for (int ks = 0; ks < 8; ++ks)
      c = wmma_bf(a[ks], ldB(W1t, 256, nt*16, ks*32, lane), c);
#pragma unroll
    for (int r = 0; r < 8; ++r){
      int row = crow + r, col = nt*16 + ccol;
      sh1[w][row][col] = (__bf16)silu_f(c[r] + b1[col]);
    }
  }
  __syncthreads();
  v16bf a0 = ldA(&sh1[w][0][0], 64, 0, lane);
  v16bf a1 = ldA(&sh1[w][0][0], 64, 32, lane);
  for (int nt = 0; nt < 4; ++nt){
    v8f c = {};
    c = wmma_bf(a0, ldB(W2t, 64, nt*16, 0,  lane), c);
    c = wmma_bf(a1, ldB(W2t, 64, nt*16, 32, lane), c);
#pragma unroll
    for (int r = 0; r < 8; ++r){
      int row = crow + r, col = nt*16 + ccol;
      int idx = (n0 + row) * 64 + col;
      Hout[idx] = H[idx] + c[r] + b2[col];
    }
  }
}

__global__ void k_xout(const float* __restrict__ X, const float* __restrict__ xupd,
                       float* __restrict__ Xout){
  int t = blockIdx.x * 256 + threadIdx.x;
  if (t < 3072){ int n = t / 3, c = t - n * 3; Xout[t] = X[t] + xupd[n*4 + c]; }
}

// ---------------------------------------------------------------- Z_out = Z + joint_z(alpha[j]*L[i]*R[j])
__global__ __launch_bounds__(128)
void k_zupd(const float* __restrict__ Z, const float* __restrict__ Lz,
            const float* __restrict__ Rz, const float* __restrict__ alphaN,
            const __bf16* __restrict__ W1tg, const __bf16* __restrict__ W2tg,
            const float* jb1, const float* jb2, float* __restrict__ Zout){
  __shared__ __bf16 sW1[64*64];
  __shared__ __bf16 sW2[64*64];
  __shared__ __bf16 sU[64][64];
  __shared__ __bf16 sh1[4][16][64];
  __shared__ float  sO[64*64];
  __shared__ float  sL[64];
  int tid = threadIdx.x, w = tid >> 5, lane = tid & 31;
  int i = blockIdx.x >> 4, j0 = (blockIdx.x & 15) << 6;
  size_t zbase = ((size_t)i * 1024 + j0) * 64;
  __builtin_prefetch(Z + zbase + (size_t)tid * 32, 0, 1);
  { // stage weights (already bf16-transposed in ws)
    const uint4* a = (const uint4*)W1tg; const uint4* b = (const uint4*)W2tg;
    uint4* da = (uint4*)sW1; uint4* db = (uint4*)sW2;
    for (int k = tid; k < 512; k += 128){ da[k] = a[k]; db[k] = b[k]; }
  }
  if (tid < 64) sL[tid] = Lz[i*64 + tid];
  __syncthreads();
  { // U[r][k] = alpha[j0+r] * L[i][k] * R[j0+r][k] in bf16
    int r = tid >> 1, h = (tid & 1) * 32;
    int j = j0 + r;
    float a = alphaN[j];
    const float* Rr = Rz + j*64 + h;
#pragma unroll
    for (int q = 0; q < 8; ++q){
      float4 v = *(const float4*)(Rr + q*4);
      unsigned p0 = pk2bf(a * sL[h+q*4+0] * v.x, a * sL[h+q*4+1] * v.y);
      unsigned p1 = pk2bf(a * sL[h+q*4+2] * v.z, a * sL[h+q*4+3] * v.w);
      *(uint2*)&sU[r][h + q*4] = make_uint2(p0, p1);
    }
  }
  __syncthreads();
  int r0 = w * 16;
  int crow = (lane >> 4) << 3, ccol = lane & 15;
  // layer1: A and all B tiles register-resident
  v16bf bw[8];
#pragma unroll
  for (int nt = 0; nt < 4; ++nt){
    bw[nt*2+0] = ldB(sW1, 64, nt*16, 0,  lane);
    bw[nt*2+1] = ldB(sW1, 64, nt*16, 32, lane);
  }
  v16bf a0 = ldA(&sU[r0][0], 64, 0, lane);
  v16bf a1 = ldA(&sU[r0][0], 64, 32, lane);
#pragma unroll
  for (int nt = 0; nt < 4; ++nt){
    v8f c = {};
    c = wmma_bf(a0, bw[nt*2+0], c);
    c = wmma_bf(a1, bw[nt*2+1], c);
#pragma unroll
    for (int r = 0; r < 8; ++r){
      int row = crow + r, col = nt*16 + ccol;
      sh1[w][row][col] = (__bf16)silu_f(c[r] + jb1[col]);
    }
  }
  __syncthreads();
  // layer2
#pragma unroll
  for (int nt = 0; nt < 4; ++nt){
    bw[nt*2+0] = ldB(sW2, 64, nt*16, 0,  lane);
    bw[nt*2+1] = ldB(sW2, 64, nt*16, 32, lane);
  }
  a0 = ldA(&sh1[w][0][0], 64, 0, lane);
  a1 = ldA(&sh1[w][0][0], 64, 32, lane);
#pragma unroll
  for (int nt = 0; nt < 4; ++nt){
    v8f c = {};
    c = wmma_bf(a0, bw[nt*2+0], c);
    c = wmma_bf(a1, bw[nt*2+1], c);
#pragma unroll
    for (int r = 0; r < 8; ++r){
      int row = crow + r, col = nt*16 + ccol;
      sO[(r0 + row) * 64 + col] = c[r] + jb2[col];
    }
  }
  __syncthreads();
  const float4* Z4 = (const float4*)(Z + zbase);
  float4* O4 = (float4*)(Zout + zbase);
  const float4* s4 = (const float4*)sO;
#pragma unroll
  for (int q = 0; q < 8; ++q){
    int idx = q * 128 + tid;
    float4 z = Z4[idx]; float4 o = s4[idx];
    o.x += z.x; o.y += z.y; o.z += z.z; o.w += z.w;
    O4[idx] = o;
  }
}

// ---------------------------------------------------------------- launch
extern "C" void kernel_launch(void* const* d_in, const int* in_sizes, int n_in,
                              void* d_out, int out_size, void* d_ws, size_t ws_size,
                              hipStream_t stream){
  (void)in_sizes; (void)n_in; (void)out_size; (void)ws_size;
  // Input order assumption: setup_inputs() insertion order; params flattened as
  // KV,Q,pair_bias,gate,phi_h,phi_x,left_z,right_z,joint_z,phi_e x (W1,b1,W2,b2).
  const float* X    = (const float*)d_in[1];
  const float* H    = (const float*)d_in[2];
  const float* E    = (const float*)d_in[3];
  const int*   eidx = (const int*)  d_in[4];
  const float* Z    = (const float*)d_in[5];
  const float *kvW1=(const float*)d_in[6],  *kvb1=(const float*)d_in[7],
              *kvW2=(const float*)d_in[8],  *kvb2=(const float*)d_in[9];
  const float *qW1 =(const float*)d_in[10], *qb1 =(const float*)d_in[11],
              *qW2 =(const float*)d_in[12], *qb2 =(const float*)d_in[13];
  const float *pbW1=(const float*)d_in[14], *pbb1=(const float*)d_in[15],
              *pbW2=(const float*)d_in[16], *pbb2=(const float*)d_in[17];
  const float *gW1 =(const float*)d_in[18], *gb1 =(const float*)d_in[19],
              *gW2 =(const float*)d_in[20], *gb2 =(const float*)d_in[21];
  const float *phW1=(const float*)d_in[22], *phb1=(const float*)d_in[23],
              *phW2=(const float*)d_in[24], *phb2=(const float*)d_in[25];
  const float *pxW1=(const float*)d_in[26], *pxb1=(const float*)d_in[27],
              *pxW2=(const float*)d_in[28], *pxb2=(const float*)d_in[29];
  const float *lW1 =(const float*)d_in[30], *lb1 =(const float*)d_in[31],
              *lW2 =(const float*)d_in[32], *lb2 =(const float*)d_in[33];
  const float *rW1 =(const float*)d_in[34], *rb1 =(const float*)d_in[35],
              *rW2 =(const float*)d_in[36], *rb2 =(const float*)d_in[37];
  const float *jW1 =(const float*)d_in[38], *jb1 =(const float*)d_in[39],
              *jW2 =(const float*)d_in[40], *jb2 =(const float*)d_in[41];
  const float *peW1=(const float*)d_in[42], *peb1=(const float*)d_in[43],
              *peW2=(const float*)d_in[44], *peb2=(const float*)d_in[45];

  float* Hout = (float*)d_out;
  float* Xout = Hout + 65536;
  float* Zout = Xout + 3072;
  float* Eout = Zout + (size_t)67108864;

  char* wsp = (char*)d_ws;
  auto alloc = [&](size_t bytes)->char*{
    char* p = wsp; wsp += (bytes + 255) & ~(size_t)255; return p;
  };
  float*    Vws    = (float*)   alloc((size_t)NE * 256 * 4);
  float*    scores = (float*)   alloc(NE * 4 * 4);
  float*    aij    = (float*)   alloc(NE * 4 * 4);
  float*    gateb  = (float*)   alloc(NE * 4 * 4);
  float*    Xrel   = (float*)   alloc(NE * 3 * 4);
  float*    coorw  = (float*)   alloc(NE * 4);
  char*     zspan0 = wsp;       // contiguous zero-init span begins
  unsigned* smaxU  = (unsigned*)alloc(NNODES * 4 * 4);
  float*    denom  = (float*)   alloc(NNODES * 4 * 4);
  float*    attnO  = (float*)   alloc(NNODES * 256 * 4);
  float*    xupd   = (float*)   alloc(NNODES * 4 * 4);
  float*    alphaN = (float*)   alloc(NNODES * 4);
  char*     zspan1 = wsp;
  float*    Lz     = (float*)   alloc(NNODES * 64 * 4);
  float*    Rz     = (float*)   alloc(NNODES * 64 * 4);
  __bf16* QW1t  = (__bf16*)alloc(64  * 64  * 2);
  __bf16* QW2t  = (__bf16*)alloc(256 * 64  * 2);
  __bf16* KVW1t = (__bf16*)alloc(128 * 64  * 2);
  __bf16* KVW2t = (__bf16*)alloc(512 * 128 * 2);
  __bf16* PXW1t = (__bf16*)alloc(64  * 256 * 2);
  __bf16* PHW1t = (__bf16*)alloc(64  * 256 * 2);
  __bf16* PHW2t = (__bf16*)alloc(64  * 64  * 2);
  __bf16* LW1t  = (__bf16*)alloc(64  * 64  * 2);
  __bf16* LW2t  = (__bf16*)alloc(64  * 64  * 2);
  __bf16* RW1t  = (__bf16*)alloc(64  * 64  * 2);
  __bf16* RW2t  = (__bf16*)alloc(64  * 64  * 2);
  __bf16* JW1t  = (__bf16*)alloc(64  * 64  * 2);
  __bf16* JW2t  = (__bf16*)alloc(64  * 64  * 2);

  int zwords = (int)((zspan1 - zspan0) / 4);
  k_zero<<<(zwords + 255) / 256, 256, 0, stream>>>((float*)zspan0, zwords);

  struct TP { const float* s; __bf16* d; int K, N; };
  TP tps[13] = {
    { qW1,       QW1t,  64, 64  }, { qW2,  QW2t,  64, 256 },
    { kvW1+128,  KVW1t, 64, 128 }, { kvW2, KVW2t, 128, 512 },
    { pxW1,      PXW1t, 256, 64 }, { phW1, PHW1t, 256, 64 },
    { phW2,      PHW2t, 64, 64  },
    { lW1, LW1t, 64, 64 }, { lW2, LW2t, 64, 64 },
    { rW1, RW1t, 64, 64 }, { rW2, RW2t, 64, 64 },
    { jW1, JW1t, 64, 64 }, { jW2, JW2t, 64, 64 },
  };
  for (int t = 0; t < 13; ++t){
    int elems = tps[t].K * tps[t].N;
    k_transpose_bf16<<<(elems + 255) / 256, 256, 0, stream>>>(
        tps[t].s, tps[t].d, tps[t].K, tps[t].N, tps[t].K, tps[t].N);
  }

  k_leftright<<<32, 64, 0, stream>>>(H, LW1t, LW2t, lb1, lb2, RW1t, RW2t, rb1, rb2, Lz, Rz);

  k_edge_main<<<NE / 32, 64, 0, stream>>>(X, H, E, eidx, Z,
      kvW1, kvb1, kvb2, qb1, qb2,
      pbW1, pbb1, pbW2, pbb2, gW1, gb1, gW2, gb2,
      QW1t, QW2t, KVW1t, KVW2t,
      Vws, scores, gateb, Xrel, smaxU);

  k_exp<<<(NE * 4 + 255) / 256, 256, 0, stream>>>(scores, smaxU, eidx, aij, denom);

  k_phix<<<NE / 32, 64, 0, stream>>>(Vws, PXW1t, pxb1, pxW2, pxb2, coorw);

  k_edge_final<<<NE / 256, 256, 0, stream>>>(eidx, denom, gateb, coorw, Xrel, E, H,
      peW1, peb1, peW2, peb2, aij, alphaN, xupd, Eout);

  k_scatter_attn<<<NE * 64 / 256, 256, 0, stream>>>(aij, Vws, eidx, attnO);

  k_phih<<<32, 64, 0, stream>>>(attnO, H, PHW1t, PHW2t, phb1, phb2, Hout);

  k_xout<<<12, 256, 0, stream>>>(X, xupd, Xout);

  k_zupd<<<16384, 128, 0, stream>>>(Z, Lz, Rz, alphaN, JW1t, JW2t, jb1, jb2, Zout);
}
